// PointDecoder4_14070312861809
// MI455X (gfx1250) — matrix operand order repaired and checked by reading, not compiled, not run on hardware
//
#include <hip/hip_runtime.h>
#include <hip/hip_bf16.h>
#include <math.h>
#include <stdint.h>

// ---------------------------------------------------------------------------
// PointDecoder (8 x 1024 x 256, 4 layers) for MI455X / gfx1250.
//  - All GEMM-class math on v_wmma_f32_16x16x32_bf16 (bf16 in, f32 accum).
//  - GEMM/attention tiles DMA'd into LDS by the Tensor Data Mover
//    (tensor_load_to_lds + s_wait_tensorcnt), descriptors per ISA ch.8.
//  - Activations keep a bf16 mirror written by producers (LN/embed/attn),
//    weights stored pre-transposed [N][K] bf16 so LDS tiles are already in
//    WMMA fragment order (no per-iteration convert/transpose VALU).
// ---------------------------------------------------------------------------

typedef __bf16 bf16;
typedef __attribute__((ext_vector_type(8)))  __bf16 v8bf;
typedef __attribute__((ext_vector_type(16))) __bf16 v16bf;
typedef __attribute__((ext_vector_type(8)))  float  v8f;
typedef unsigned int v4u __attribute__((ext_vector_type(4)));
typedef int v8i __attribute__((ext_vector_type(8)));
typedef int v4i __attribute__((ext_vector_type(4)));

#define BSZ   8
#define NHD   8
#define DH    32
#define HD    256
#define PP    1024
#define PARTN 4
#define NL    4
#define FFD   1024
#define ROWS  (BSZ*PP)   // 8192

__device__ __forceinline__ bf16 f2bf(float f) {
  union { float f; unsigned u; } x; x.f = f;
  unsigned r = x.u + 0x7FFFu + ((x.u >> 16) & 1u);   // round-to-nearest-even
  unsigned short h = (unsigned short)(r >> 16);
  return __builtin_bit_cast(bf16, h);
}

// ---------------------------------------------------------------------------
// Tensor Data Mover: 2D tile (bf16) global -> LDS.
//  D# group0: count=1 | lds_addr | global_addr[56:0] | type=2
//  D# group1: data_size=1(2B), tensor_dim0/1 (OOB), tile_dim0/1, dim0_stride
//  tile lands row-major [tile_h][tile_w] at lds_addr.
//  clang-23 lane: 6-arg builtin (g0, g1, g2, g3, extra v8i, cpol).
// ---------------------------------------------------------------------------
__device__ __forceinline__ void tdm_load_2d(unsigned lds_addr, const void* g,
                                            unsigned tile_w, unsigned tile_h,
                                            unsigned long long stride_elems,
                                            unsigned dim0, unsigned dim1) {
  unsigned long long ga = (unsigned long long)(uintptr_t)g;
  v4u g0;
  g0[0] = 1u;                                          // count=1, no gather
  g0[1] = lds_addr;                                    // LDS byte address
  g0[2] = (unsigned)(ga & 0xFFFFFFFFu);                // global_addr[31:0]
  g0[3] = (unsigned)((ga >> 32) & 0x01FFFFFFu) | (2u << 30);  // [56:32]|type=2
  v8i g1;
  g1[0] = (int)(1u << 16);                             // wg_mask=0, data_size=1
  g1[1] = (int)((dim0 & 0xFFFFu) << 16);               // tensor_dim0[15:0]@48
  g1[2] = (int)((dim0 >> 16) | ((dim1 & 0xFFFFu) << 16));
  g1[3] = (int)((dim1 >> 16) | (tile_w << 16));        // tile_dim0@112
  g1[4] = (int)(tile_h);                               // tile_dim1, tile_dim2=0
  g1[5] = (int)(stride_elems & 0xFFFFFFFFu);           // dim0_stride[31:0]@160
  g1[6] = (int)((stride_elems >> 32) & 0xFFFFu);       // dim0_stride[47:32]
  g1[7] = 0;
  v4i gz4 = {0, 0, 0, 0};
  v8i gz8 = {0, 0, 0, 0, 0, 0, 0, 0};
  __builtin_amdgcn_tensor_load_to_lds(g0, g1, gz4, gz4, gz8, 0);
}
__device__ __forceinline__ unsigned lds_off(const void* p) {
  return (unsigned)(uintptr_t)p;   // LDS aperture: addr[31:0] = LDS offset
}

// --- WMMA fragment loaders: A and B both use contiguous k = 16*half + s -----
__device__ __forceinline__ v16bf load_a_frag(const bf16* base, int stride,
                                             int row_base, int lane) {
  const int half = lane >> 4;
  const bf16* p = base + (size_t)(row_base + (lane & 15)) * stride;
  v8bf lo = *(const v8bf*)(p + half * 16);
  v8bf hi = *(const v8bf*)(p + 8 + half * 16);
  v16bf a;
#pragma unroll
  for (int j = 0; j < 8; j++) { a[j] = lo[j]; a[j + 8] = hi[j]; }
  return a;
}
__device__ __forceinline__ v16bf load_b_frag(const bf16* base, int stride,
                                             int col_base, int lane) {
  const int half = lane >> 4;
  const bf16* p = base + (size_t)(col_base + (lane & 15)) * stride + half * 16;
  return *(const v16bf*)p;
}

// ---------------------------------------------------------------------------
// weight conversion: f32 [K][N] -> bf16 transposed [N][K]
// ---------------------------------------------------------------------------
__global__ void k_cvt_t(const float* __restrict__ s, bf16* __restrict__ d,
                        int K, int N) {
  for (int i = blockIdx.x * 256 + threadIdx.x; i < K * N; i += gridDim.x * 256) {
    int k = i / N, n = i % N;
    d[(size_t)n * K + k] = f2bf(s[i]);
  }
}

// ---------------------------------------------------------------------------
// _build: counting sort by label + slot assignment (one block per batch)
// ---------------------------------------------------------------------------
__global__ __launch_bounds__(256) void k_build(
    const float* __restrict__ ipt, const int* __restrict__ ilab,
    const float* __restrict__ irat, const float* __restrict__ prat,
    const int* __restrict__ nvec, const float* __restrict__ tmpl,
    float* __restrict__ balls, int* __restrict__ labels,
    int* __restrict__ issel) {
  __shared__ int cnt[PARTN], sel[PARTN], ends[PARTN], starts[PARTN],
      lstart[PARTN], s_total;
  __shared__ float prob[PARTN];
  __shared__ int order[PP];
  const int b = blockIdx.x, t = threadIdx.x;
  if (t < PARTN) cnt[t] = 0;
  __syncthreads();
  for (int i = t; i < PP; i += 256) atomicAdd(&cnt[ilab[b * PP + i]], 1);
  __syncthreads();
  if (t < PARTN) {
    float pp = roundf(prat[b * PARTN + t] * (float)PP);
    float pn = roundf(prat[b * PARTN + t] * (float)PP -
                      irat[b * PARTN + t] * (float)nvec[b]);
    float pr = fmaxf(pn, 0.f);
    prob[t] = pr;
    int sv = (int)(pp - pr);                 // astype(int32) truncation
    sv = sv < 0 ? 0 : sv;
    sv = sv > cnt[t] ? cnt[t] : sv;
    sel[t] = sv;
  }
  __syncthreads();
  if (t == 0) {
    int remaining = PP;
    float psum = 0.f;
    for (int q = 0; q < PARTN; q++) { remaining -= sel[q]; psum += prob[q]; }
    bool has_pad = (remaining > 0) && (psum >= 1.0f);
    int pad[PARTN]; int pbsum = 0;
    for (int q = 0; q < PARTN; q++) {
      float pf = (float)remaining * prob[q] / fmaxf(psum, 1e-9f);
      pad[q] = (int)floorf(pf);
      pbsum += pad[q];
    }
    int amax = 0;
    for (int q = 1; q < PARTN; q++) if (prob[q] > prob[amax]) amax = q;
    pad[amax] += remaining - pbsum;
    int e = 0;
    for (int q = 0; q < PARTN; q++) {
      int pq = has_pad ? (pad[q] < 0 ? 0 : (pad[q] > PP ? PP : pad[q])) : 0;
      int pn = sel[q] + pq;
      starts[q] = e; e += pn; ends[q] = e;
    }
    s_total = ends[PARTN - 1];
    int ls = 0;
    for (int q = 0; q < PARTN; q++) { lstart[q] = ls; ls += cnt[q]; }
  }
  __syncthreads();
  if (t < PARTN) {                  // stable counting sort: order = argsort(label)
    int pos = lstart[t];
    for (int i = 0; i < PP; i++)
      if (ilab[b * PP + i] == t) order[pos++] = i;
  }
  __syncthreads();
  for (int p = t; p < PP; p += 256) {
    int j = 0;
    for (int q = 0; q < PARTN; q++) j += (p >= ends[q]) ? 1 : 0;
    if (j > PARTN - 1) j = PARTN - 1;
    bool valid = p < s_total;
    int off = p - starts[j];
    int sj = sel[j];
    bool is_sel = valid && (off < sj);
    float x = 0.f, y = 0.f, z = 0.f;
    if (is_sel) {
      int sp = lstart[j] + off;
      sp = sp < 0 ? 0 : (sp > PP - 1 ? PP - 1 : sp);
      int src = order[sp];
      const float* q3 = &ipt[((size_t)b * PP + src) * 3];
      x = q3[0]; y = q3[1]; z = q3[2];
    } else if (valid) {
      int po = off - sj;
      po = po < 0 ? 0 : (po > PP - 1 ? PP - 1 : po);
      const float* q3 = &tmpl[((size_t)j * PP + po) * 3];
      x = q3[0]; y = q3[1]; z = q3[2];
    }
    size_t r = (size_t)b * PP + p;
    balls[r * 3 + 0] = x; balls[r * 3 + 1] = y; balls[r * 3 + 2] = z;
    labels[r] = valid ? j : 0;
    issel[r] = is_sel ? 1 : 0;
  }
}

// ---------------------------------------------------------------------------
// mem = encoder_output_full[:, PART:, :], contiguized + bf16
// ---------------------------------------------------------------------------
__global__ __launch_bounds__(256) void k_copy_mem(const float* __restrict__ eof,
                                                  bf16* __restrict__ memb) {
  int row = blockIdx.x, t = threadIdx.x;
  int b = row >> 10, p = row & 1023;
  memb[(size_t)row * HD + t] =
      f2bf(eof[((size_t)b * (PARTN + PP) + PARTN + p) * HD + t]);
}

// ---------------------------------------------------------------------------
// x = concat(balls, enc[label], one_hot(is_sel)) @ Wemb + bemb + label_emb
// writes f32 X (residual) + bf16 mirror (GEMM feed)
// ---------------------------------------------------------------------------
__global__ __launch_bounds__(256) void k_embed(
    const float* __restrict__ enc, const float* __restrict__ W,
    const float* __restrict__ bias, const float* __restrict__ lemb,
    const float* __restrict__ balls, const int* __restrict__ labels,
    const int* __restrict__ issel, float* __restrict__ X,
    bf16* __restrict__ Xb) {
  __shared__ float pf[HD];
  int row = blockIdx.x, t = threadIdx.x;
  int b = row >> 10;
  int lab = labels[row];
  pf[t] = enc[((size_t)b * PARTN + lab) * HD + t];
  __syncthreads();
  float acc = bias[t] + lemb[(size_t)lab * HD + t];
  float b0 = balls[(size_t)row * 3 + 0], b1 = balls[(size_t)row * 3 + 1],
        b2 = balls[(size_t)row * 3 + 2];
  acc += b0 * W[0 * HD + t] + b1 * W[1 * HD + t] + b2 * W[2 * HD + t];
#pragma unroll 8
  for (int k = 0; k < HD; k++) acc += pf[k] * W[(3 + k) * HD + t];
  float r1 = (float)issel[row];
  acc += (1.f - r1) * W[259 * HD + t] + r1 * W[260 * HD + t];
  X[(size_t)row * HD + t] = acc;
  Xb[(size_t)row * HD + t] = f2bf(acc);
}

// ---------------------------------------------------------------------------
// WMMA GEMM: A[M,K] bf16 @ Wt[N,K] bf16 (pre-transposed) + bias.
// Tiles DMA'd to LDS by the TDM (issued by wave 0, published via barrier).
// Block tile 128x64, BK=32, 8 waves, each wave a 32x32 subtile (2x2 WMMA).
// Outputs: optional f32 C and/or bf16 Cb mirror.
// ---------------------------------------------------------------------------
__global__ __launch_bounds__(256) void k_gemm(
    const bf16* __restrict__ A, const bf16* __restrict__ Wt,
    const float* __restrict__ bias, float* __restrict__ C,
    bf16* __restrict__ Cb, int M, int N, int K, int relu) {
  __shared__ bf16 sA[128][32];
  __shared__ bf16 sB[64][32];   // [n][k] (Wt is already [N][K])
  const int tid = threadIdx.x, lane = tid & 31, wave = tid >> 5;
  const int bm = blockIdx.x * 128;
  const int bn = blockIdx.y * 64;
  const int wr = (wave >> 1) * 32;
  const int wc = (wave & 1) * 32;
  const unsigned ldsA = lds_off(&sA[0][0]);
  const unsigned ldsB = lds_off(&sB[0][0]);
  v8f acc[2][2] = {};
  for (int k0 = 0; k0 < K; k0 += 32) {
    if (wave == 0) {
      tdm_load_2d(ldsA, A + (size_t)bm * K + k0, 32, 128,
                  (unsigned long long)K, 32, 128);
      tdm_load_2d(ldsB, Wt + (size_t)bn * K + k0, 32, 64,
                  (unsigned long long)K, 32, (unsigned)(N - bn)); // OOB rows->0
      __builtin_amdgcn_s_wait_tensorcnt(0);
    }
    __syncthreads();
    v16bf af0 = load_a_frag(&sA[0][0], 32, wr, lane);
    v16bf af1 = load_a_frag(&sA[0][0], 32, wr + 16, lane);
    v16bf bf0 = load_b_frag(&sB[0][0], 32, wc, lane);
    v16bf bf1 = load_b_frag(&sB[0][0], 32, wc + 16, lane);
    acc[0][0] = __builtin_amdgcn_wmma_f32_16x16x32_bf16(false, af0, false, bf0, (short)0, acc[0][0], false, false);
    acc[0][1] = __builtin_amdgcn_wmma_f32_16x16x32_bf16(false, af0, false, bf1, (short)0, acc[0][1], false, false);
    acc[1][0] = __builtin_amdgcn_wmma_f32_16x16x32_bf16(false, af1, false, bf0, (short)0, acc[1][0], false, false);
    acc[1][1] = __builtin_amdgcn_wmma_f32_16x16x32_bf16(false, af1, false, bf1, (short)0, acc[1][1], false, false);
    __syncthreads();
  }
  const int half = lane >> 4, ln = lane & 15;
#pragma unroll
  for (int i = 0; i < 2; i++)
#pragma unroll
    for (int j = 0; j < 2; j++) {
      int col = bn + wc + 16 * j + ln;
      if (col >= N) continue;
      float bv = bias ? bias[col] : 0.f;
#pragma unroll
      for (int r = 0; r < 8; r++) {             // C layout: row = base+r+8*half
        int row = bm + wr + 16 * i + r + 8 * half;
        float v = acc[i][j][r] + bv;
        if (relu) v = fmaxf(v, 0.f);
        if (C)  C[(size_t)row * N + col] = v;
        if (Cb) Cb[(size_t)row * N + col] = f2bf(v);
      }
    }
}

// ---------------------------------------------------------------------------
// Fused flash attention per (b, h, 64-query tile); Tk = 1024, mask all-true.
// Q/K tiles via TDM (2D strided tile, stride=HD); V transposed manually.
// S via wmma (K = DH = 32); O accumulation via wmma over P*V. Output bf16.
// ---------------------------------------------------------------------------
__global__ __launch_bounds__(256) void k_attn(
    const bf16* __restrict__ Q, const bf16* __restrict__ Kx,
    const bf16* __restrict__ V, bf16* __restrict__ O) {
  __shared__ bf16 sQ[64][32];
  __shared__ bf16 sK[64][32];
  __shared__ bf16 sVt[32][64];                  // [dh][key]
  __shared__ float sS[64][64];
  __shared__ bf16 sP[64][64];
  __shared__ float sM[64], sL[64], sAl[64];
  const int tid = threadIdx.x, lane = tid & 31, wave = tid >> 5;
  const int b = blockIdx.x / NHD, h = blockIdx.x % NHD;
  const int q0 = blockIdx.y * 64;
  const size_t rowbase = (size_t)b * PP;
  const int hc = h * DH;
  if (wave == 0) {                              // Q tile DMA, once
    tdm_load_2d(lds_off(&sQ[0][0]), Q + (rowbase + q0) * HD + hc,
                32, 64, (unsigned long long)HD, 32, 64);
  }
  if (tid < 64) { sM[tid] = -1e30f; sL[tid] = 0.f; }
  const int ro = (wave >> 1) * 16, co = (wave & 1) * 16;  // O tile (16x16)
  const int sr = (wave >> 1) * 16, sc = (wave & 1) * 32;  // S tile (16x32)
  const float scale = 0.17677669529663687f;               // 1/sqrt(DH)
  v8f oacc = {};
  for (int kt = 0; kt < PP / 64; kt++) {
    const int kb = kt * 64;
    if (wave == 0) {                            // K tile DMA
      tdm_load_2d(lds_off(&sK[0][0]), Kx + (rowbase + kb) * HD + hc,
                  32, 64, (unsigned long long)HD, 32, 64);
    }
    {                                           // V tile, transposed by hand
      int r = tid >> 2, c8 = (tid & 3) * 8;
      v8bf vv = *(const v8bf*)&V[(rowbase + kb + r) * HD + hc + c8];
#pragma unroll
      for (int j = 0; j < 8; j++) sVt[c8 + j][r] = vv[j];
    }
    if (wave == 0) __builtin_amdgcn_s_wait_tensorcnt(0);
    __syncthreads();
    // S = Q K^T
    {
      v16bf aq = load_a_frag(&sQ[0][0], 32, sr, lane);
      const int half = lane >> 4, ln = lane & 15;
#pragma unroll
      for (int j = 0; j < 2; j++) {
        v16bf bk = load_b_frag(&sK[0][0], 32, sc + 16 * j, lane);
        v8f s = {};
        s = __builtin_amdgcn_wmma_f32_16x16x32_bf16(false, aq, false, bk, (short)0, s, false, false);
#pragma unroll
        for (int r = 0; r < 8; r++)
          sS[sr + r + 8 * half][sc + 16 * j + ln] = s[r] * scale;
      }
    }
    __syncthreads();
    if (tid < 64) {                             // online softmax, one row/thread
      float mo = sM[tid], mt = -1e30f;
      for (int c = 0; c < 64; c++) mt = fmaxf(mt, sS[tid][c]);
      float mn = fmaxf(mo, mt);
      float al = __expf(mo - mn);
      float sum = 0.f;
      for (int c = 0; c < 64; c++) {
        float p = __expf(sS[tid][c] - mn);
        sP[tid][c] = f2bf(p);
        sum += p;
      }
      sL[tid] = sL[tid] * al + sum;
      sM[tid] = mn;
      sAl[tid] = al;
    }
    __syncthreads();
    {
      const int half = lane >> 4;
#pragma unroll
      for (int r = 0; r < 8; r++) oacc[r] *= sAl[ro + r + 8 * half];
#pragma unroll
      for (int ch = 0; ch < 2; ch++) {          // O += P(64x64) @ V(64x32)
        v16bf ap = load_a_frag(&sP[0][0] + ch * 32, 64, ro, lane);
        v16bf bv = load_b_frag(&sVt[0][0] + ch * 32, 64, co, lane);
        oacc = __builtin_amdgcn_wmma_f32_16x16x32_bf16(false, ap, false, bv, (short)0, oacc, false, false);
      }
    }
    __syncthreads();
  }
  const int half = lane >> 4, ln = lane & 15;
#pragma unroll
  for (int r = 0; r < 8; r++) {
    int row = ro + r + 8 * half;
    O[(rowbase + q0 + row) * HD + hc + co + ln] =
        f2bf(oacc[r] / fmaxf(sL[row], 1e-20f));
  }
}

// ---------------------------------------------------------------------------
// out = LN(a (+ bres)) [*scale+bias] [relu] ; one row per block, D <= 256
// optional bf16 mirror for GEMM consumers
// ---------------------------------------------------------------------------
__global__ __launch_bounds__(256) void k_add_ln(
    const float* __restrict__ a, const float* __restrict__ bres,
    const float* __restrict__ sc, const float* __restrict__ bi,
    float* __restrict__ out, bf16* __restrict__ outb, int D, int relu) {
  __shared__ float red[256];
  int row = blockIdx.x, t = threadIdx.x;
  float v = 0.f;
  if (t < D) {
    v = a[(size_t)row * D + t];
    if (bres) v += bres[(size_t)row * D + t];
  }
  red[t] = v; __syncthreads();
  for (int s = 128; s > 0; s >>= 1) { if (t < s) red[t] += red[t + s]; __syncthreads(); }
  float mean = red[0] / D; __syncthreads();
  float dv = (t < D) ? (v - mean) : 0.f;
  red[t] = dv * dv; __syncthreads();
  for (int s = 128; s > 0; s >>= 1) { if (t < s) red[t] += red[t + s]; __syncthreads(); }
  float var = red[0] / D;
  float y = dv * rsqrtf(var + 1e-5f);
  if (sc) y = y * sc[t] + bi[t];
  if (relu) y = fmaxf(y, 0.f);
  if (t < D) {
    if (out)  out[(size_t)row * D + t] = y;
    if (outb) outb[(size_t)row * D + t] = f2bf(y);
  }
}

// ---------------------------------------------------------------------------
// head: y3 = y2(32) @ w3(32x3) + b3 + balls -> d_out[:24576]; labels appended
// ---------------------------------------------------------------------------
__global__ void k_final(const float* __restrict__ y2, const float* __restrict__ w3,
                        const float* __restrict__ b3, const float* __restrict__ balls,
                        float* __restrict__ out) {
  int id = blockIdx.x * 256 + threadIdx.x;
  if (id >= ROWS * 3) return;
  int row = id / 3, c = id % 3;
  float s = b3[c];
#pragma unroll
  for (int k = 0; k < 32; k++) s += y2[(size_t)row * 32 + k] * w3[k * 3 + c];
  out[id] = s + balls[id];
}
__global__ void k_labels(const int* __restrict__ lab, float* __restrict__ out) {
  int id = blockIdx.x * 256 + threadIdx.x;
  if (id < ROWS) out[id] = (float)lab[id];
}

// ---------------------------------------------------------------------------
// Host orchestration. Input order: setup_inputs() insertion order; params
// expanded via jax tree_leaves order (recursive alphabetical):
//  0 encoder_output 1 input_point 2 input_label 3 input_ratio 4 point_ratio
//  5 n 6 encoder_output_full 7 mask(unused,all true) 8 template
//  9 embed_b 10 embed_w 11 f.b1 12 f.b2 13 f.b3 14 f.w1 15 f.w2 16 f.w3
//  17 label_emb, then 4 layers x 26:
//   cross{bk,bo,bq,bv,wk,wo,wq,wv}(+0..7) ff_b1(+8) ff_b2(+9) ff_w1(+10)
//   ff_w2(+11) ln1_b(+12) ln1_s(+13) ln2_b(+14) ln2_s(+15) ln3_b(+16)
//   ln3_s(+17) self{bk,bo,bq,bv,wk,wo,wq,wv}(+18..25)
// ---------------------------------------------------------------------------
extern "C" void kernel_launch(void* const* d_in, const int* in_sizes, int n_in,
                              void* d_out, int out_size, void* d_ws,
                              size_t ws_size, hipStream_t stream) {
  (void)in_sizes; (void)n_in; (void)out_size; (void)ws_size;
  const float* enc   = (const float*)d_in[0];
  const float* ipt   = (const float*)d_in[1];
  const int*   ilab  = (const int*)d_in[2];
  const float* irat  = (const float*)d_in[3];
  const float* prat  = (const float*)d_in[4];
  const int*   nvec  = (const int*)d_in[5];
  const float* eof   = (const float*)d_in[6];
  const float* tmpl  = (const float*)d_in[8];
  const float* emb_b = (const float*)d_in[9];
  const float* emb_w = (const float*)d_in[10];
  const float* f_b1  = (const float*)d_in[11];
  const float* f_b2  = (const float*)d_in[12];
  const float* f_b3  = (const float*)d_in[13];
  const float* f_w1  = (const float*)d_in[14];
  const float* f_w2  = (const float*)d_in[15];
  const float* f_w3  = (const float*)d_in[16];
  const float* lemb  = (const float*)d_in[17];
  auto Lp = [&](int l, int k) -> const float* {
    return (const float*)d_in[18 + l * 26 + k];
  };

  // ---- workspace carve ----
  char* wsp = (char*)d_ws;
  size_t off = 0;
  auto alloc = [&](size_t bytes) -> char* {
    char* p = wsp + off;
    off += (bytes + 255) & ~(size_t)255;
    return p;
  };
  const size_t WB_ELEMS =
      (size_t)NL * (8 * HD * HD + 2 * HD * FFD) + HD * 128 + 128 * 32;
  bf16* WB   = (bf16*)alloc(WB_ELEMS * sizeof(bf16));   // transposed weights
  float* X   = (float*)alloc((size_t)ROWS * HD * 4);
  float* T0  = (float*)alloc((size_t)ROWS * HD * 4);
  bf16* Xb   = (bf16*)alloc((size_t)ROWS * HD * 2);
  bf16* Qbb  = (bf16*)alloc((size_t)ROWS * HD * 2);
  bf16* Kbb  = (bf16*)alloc((size_t)ROWS * HD * 2);
  bf16* Vbb  = (bf16*)alloc((size_t)ROWS * HD * 2);
  bf16* AOb  = (bf16*)alloc((size_t)ROWS * HD * 2);
  bf16* MEMb = (bf16*)alloc((size_t)ROWS * HD * 2);
  bf16* FFbb = (bf16*)alloc((size_t)ROWS * FFD * 2);
  float* BALLS = (float*)alloc((size_t)ROWS * 3 * 4);
  int*   LAB   = (int*)alloc((size_t)ROWS * 4);
  int*   ISEL  = (int*)alloc((size_t)ROWS * 4);
  float* Y1    = (float*)alloc((size_t)ROWS * 128 * 4);
  bf16*  Y1b   = (bf16*)alloc((size_t)ROWS * 128 * 2);
  float* Y2    = (float*)alloc((size_t)ROWS * 32 * 4);

  bf16* w_self[NL][4];  // wq, wk, wv, wo  (all [N][K] transposed bf16)
  bf16* w_cross[NL][4];
  bf16* w_ff1[NL];
  bf16* w_ff2[NL];
  {
    bf16* cur = WB;
    for (int l = 0; l < NL; l++) {
      for (int m = 0; m < 4; m++) { w_self[l][m] = cur; cur += HD * HD; }
      for (int m = 0; m < 4; m++) { w_cross[l][m] = cur; cur += HD * HD; }
      w_ff1[l] = cur; cur += HD * FFD;
      w_ff2[l] = cur; cur += FFD * HD;
    }
  }
  bf16* fw1b = WB + (size_t)NL * (8 * HD * HD + 2 * HD * FFD);
  bf16* fw2b = fw1b + HD * 128;

  auto cvt = [&](const float* s, bf16* d, int K, int N) {
    int g = (K * N + 255) / 256;
    if (g > 4096) g = 4096;
    k_cvt_t<<<g, 256, 0, stream>>>(s, d, K, N);
  };
  for (int l = 0; l < NL; l++) {
    cvt(Lp(l, 24), w_self[l][0], HD, HD);    // self.wq
    cvt(Lp(l, 22), w_self[l][1], HD, HD);    // self.wk
    cvt(Lp(l, 25), w_self[l][2], HD, HD);    // self.wv
    cvt(Lp(l, 23), w_self[l][3], HD, HD);    // self.wo
    cvt(Lp(l, 6),  w_cross[l][0], HD, HD);   // cross.wq
    cvt(Lp(l, 4),  w_cross[l][1], HD, HD);   // cross.wk
    cvt(Lp(l, 7),  w_cross[l][2], HD, HD);   // cross.wv
    cvt(Lp(l, 5),  w_cross[l][3], HD, HD);   // cross.wo
    cvt(Lp(l, 10), w_ff1[l], HD, FFD);
    cvt(Lp(l, 11), w_ff2[l], FFD, HD);
  }
  cvt(f_w1, fw1b, HD, 128);
  cvt(f_w2, fw2b, 128, 32);

  auto gemm = [&](const bf16* A, const bf16* Wt, const float* bias, float* C,
                  bf16* Cb, int M, int N, int K, int relu) {
    dim3 g(M / 128, (N + 63) / 64);
    k_gemm<<<g, 256, 0, stream>>>(A, Wt, bias, C, Cb, M, N, K, relu);
  };
  const dim3 attn_grid(BSZ * NHD, PP / 64);

  // ---- pipeline ----
  k_build<<<BSZ, 256, 0, stream>>>(ipt, ilab, irat, prat, nvec, tmpl,
                                   BALLS, LAB, ISEL);
  k_copy_mem<<<ROWS, 256, 0, stream>>>(eof, MEMb);
  k_embed<<<ROWS, 256, 0, stream>>>(enc, emb_w, emb_b, lemb, BALLS, LAB, ISEL,
                                    X, Xb);

  for (int l = 0; l < NL; l++) {
    // self-attention block
    gemm(Xb, w_self[l][0], Lp(l, 20), nullptr, Qbb, ROWS, HD, HD, 0);
    gemm(Xb, w_self[l][1], Lp(l, 18), nullptr, Kbb, ROWS, HD, HD, 0);
    gemm(Xb, w_self[l][2], Lp(l, 21), nullptr, Vbb, ROWS, HD, HD, 0);
    k_attn<<<attn_grid, 256, 0, stream>>>(Qbb, Kbb, Vbb, AOb);
    gemm(AOb, w_self[l][3], Lp(l, 19), T0, nullptr, ROWS, HD, HD, 0);
    k_add_ln<<<ROWS, 256, 0, stream>>>(X, T0, Lp(l, 13), Lp(l, 12), X, Xb, HD, 0);
    // cross-attention block (mask all-true)
    gemm(Xb, w_cross[l][0], Lp(l, 2), nullptr, Qbb, ROWS, HD, HD, 0);
    gemm(MEMb, w_cross[l][1], Lp(l, 0), nullptr, Kbb, ROWS, HD, HD, 0);
    gemm(MEMb, w_cross[l][2], Lp(l, 3), nullptr, Vbb, ROWS, HD, HD, 0);
    k_attn<<<attn_grid, 256, 0, stream>>>(Qbb, Kbb, Vbb, AOb);
    gemm(AOb, w_cross[l][3], Lp(l, 1), T0, nullptr, ROWS, HD, HD, 0);
    k_add_ln<<<ROWS, 256, 0, stream>>>(X, T0, Lp(l, 15), Lp(l, 14), X, Xb, HD, 0);
    // feed-forward
    gemm(Xb, w_ff1[l], Lp(l, 8), nullptr, FFbb, ROWS, FFD, HD, 1);
    gemm(FFbb, w_ff2[l], Lp(l, 9), T0, nullptr, ROWS, HD, FFD, 0);
    k_add_ln<<<ROWS, 256, 0, stream>>>(X, T0, Lp(l, 17), Lp(l, 16), X, Xb, HD, 0);
  }

  // head
  gemm(Xb, fw1b, f_b1, Y1, nullptr, ROWS, 128, HD, 0);
  k_add_ln<<<ROWS, 256, 0, stream>>>(Y1, nullptr, nullptr, nullptr, nullptr,
                                     Y1b, 128, 1);
  gemm(Y1b, fw2b, f_b2, Y2, nullptr, ROWS, 32, 128, 0);
  k_add_ln<<<ROWS, 256, 0, stream>>>(Y2, nullptr, nullptr, nullptr, Y2, nullptr,
                                     32, 1);
  k_final<<<(ROWS * 3 + 255) / 256, 256, 0, stream>>>(Y2, f_w3, f_b3, BALLS,
                                                      (float*)d_out);
  k_labels<<<(ROWS + 255) / 256, 256, 0, stream>>>(LAB, (float*)d_out + ROWS * 3);
}
